// LightHyperFormerConv_45621142618208
// MI455X (gfx1250) — matrix operand order repaired and checked by reading, not compiled, not run on hardware
//
#include <hip/hip_runtime.h>
#include <hip/hip_bf16.h>

typedef __attribute__((ext_vector_type(16))) _Float16 v16h;
typedef __attribute__((ext_vector_type(8)))  _Float16 v8h;
typedef __attribute__((ext_vector_type(8)))  float    v8f;
typedef __attribute__((ext_vector_type(4)))  unsigned int u32x4;
typedef __attribute__((ext_vector_type(8)))  int      i32x8;
typedef __attribute__((ext_vector_type(4)))  int      i32x4;
typedef _Float16 half_t;

#define NN   32768          // nodes
#define NE   65536          // edges
#define NH   8              // heads
#define NRF  128            // random features
#define GR   (NH * NN)      // 262144 flat rows
#define EPSF 1e-7f
#define INV_SQRT63 0.12598815766974242f
#define SQRT2F     1.4142135623730951f

// ---- CDNA5 TDM availability (device pass only) ----
#ifdef __AMDGCN__
#if defined(__has_builtin)
#if __has_builtin(__builtin_amdgcn_tensor_load_to_lds) && \
    __has_builtin(__builtin_amdgcn_s_wait_tensorcnt)
#define USE_TDM 1
#endif
#endif
#ifndef USE_TDM
#warning "CDNA5 TDM builtins unavailable: k4_last falls back to direct loads"
#endif
#endif
#ifndef USE_TDM
#define USE_TDM 0
#endif

// ---- workspace layout (bytes) ----
static constexpr size_t OFF_U   = 0;                          // u rows f16  [NN][64]
static constexpr size_t SZ_U    = (size_t)NN * 64 * 2;
static constexpr size_t OFF_QF  = OFF_U  + SZ_U;              // phi(q) f16  [GR][128]
static constexpr size_t SZ_F    = (size_t)GR * 128 * 2;
static constexpr size_t OFF_KF  = OFF_QF + SZ_F;              // phi(k) f16  [GR][128]
static constexpr size_t OFF_VT  = OFF_KF + SZ_F;              // vT f16      [8][64][32768]
static constexpr size_t SZ_VT   = (size_t)GR * 64 * 2;
static constexpr size_t OFF_L32 = OFF_VT + SZ_VT;             // lastT f32   [8][64][128]
static constexpr size_t SZ_L32  = (size_t)8 * 64 * 128 * 4;
static constexpr size_t OFF_L16 = OFF_L32 + SZ_L32;           // lastT f16   [8][64][128]
static constexpr size_t SZ_L16  = (size_t)8 * 64 * 128 * 2;
static constexpr size_t OFF_T   = OFF_L16 + SZ_L16;           // t accum f32 [NN][64]
static constexpr size_t SZ_T    = (size_t)NN * 64 * 4;
static constexpr size_t OFF_WP  = OFF_T + SZ_T;               // W frag-packed f16
static constexpr size_t SZ_WP   = (size_t)24 * 2 * 4 * 32 * 16 * 2;   // 196608
static constexpr size_t OFF_PP  = OFF_WP + SZ_WP;             // Ppad frag-packed f16
static constexpr size_t SZ_PP   = (size_t)2 * 8 * 32 * 16 * 2;        // 16384

__device__ __forceinline__ v8f wmma_f16(v16h a, v16h b, v8f c) {
  return __builtin_amdgcn_wmma_f32_16x16x32_f16(false, a, false, b, (short)0, c,
                                                false, false);
}
// A fragment: 16 contiguous halves (K 16 per lane half)
__device__ __forceinline__ v16h load_a16(const half_t* base) {
  v8h lo = *(const v8h*)base;
  v8h hi = *(const v8h*)(base + 8);
  v16h a;
  #pragma unroll
  for (int j = 0; j < 8; ++j) { a[j] = lo[j]; a[8 + j] = hi[j]; }
  return a;
}

// ---------------- K0: pack W / Ppad into per-lane WMMA-B fragment order -----
// Wpack[seg][ks][nt][lane][j] = W_seg[o = nt*16+(lane&15)][c = ks*32+16*(lane>>4)+j]
__global__ void __launch_bounds__(256) k0_pack(const float* __restrict__ Wq,
                                               const float* __restrict__ Wk,
                                               const float* __restrict__ Wv,
                                               const float* __restrict__ P,
                                               half_t* __restrict__ Wpack,
                                               half_t* __restrict__ Ppack) {
  int tid = blockIdx.x * blockDim.x + threadIdx.x;
  if (tid < 24 * 2 * 4 * 32 * 16) {
    int j    = tid & 15;
    int lane = (tid >> 4) & 31;
    int nt   = (tid >> 9) & 3;
    int ks   = (tid >> 11) & 1;
    int seg  = tid >> 12;
    int typ = seg >> 3, h = seg & 7;
    const float* W = ((typ == 0) ? Wq : (typ == 1) ? Wk : Wv) + (size_t)h * 4096;
    int o = nt * 16 + (lane & 15);
    int c = ks * 32 + 16 * (lane >> 4) + j;
    Wpack[tid] = (half_t)W[o * 64 + c];
  }
  int t2 = tid - 24 * 2 * 4 * 32 * 16;
  if (t2 >= 0 && t2 < 2 * 8 * 32 * 16) {
    int j    = t2 & 15;
    int lane = (t2 >> 4) & 31;
    int nt   = (t2 >> 9) & 7;
    int ks   = t2 >> 12;
    int nf = nt * 16 + (lane & 15);
    int k  = ks * 32 + 16 * (lane >> 4) + j;
    Ppack[t2] = (half_t)((k >= 1) ? P[(size_t)(k - 1) * NRF + nf] : 0.0f);
  }
}

// ---------------- K1: u = logmap0(proj(expmap0(z))) , one wave per node -----
__global__ void __launch_bounds__(256) k1_build_u(const float* __restrict__ z,
                                                  half_t* __restrict__ u_h) {
  int node = (int)((blockIdx.x * blockDim.x + threadIdx.x) >> 5);
  int lane = threadIdx.x & 31;
  const float* zr = z + (size_t)node * 64;
  float z0 = zr[lane];
  float z1 = zr[lane + 32];
  float s = ((lane == 0) ? 0.f : z0 * z0) + z1 * z1;   // sum over ch 1..63
  #pragma unroll
  for (int m = 16; m; m >>= 1) s += __shfl_xor(s, m, 32);
  float xn    = fmaxf(sqrtf(s), EPSF);
  float sh    = sinhf(xn);
  float scale = sh / xn;
  float ss2   = scale * scale * s;
  float t     = sqrtf(1.0f + ss2);
  float yn    = fmaxf(sqrtf(ss2), EPSF);
  float tt    = fmaxf(t, 1.0f + EPSF);
  float ac    = acoshf(tt) / yn;
  float r0    = (lane == 0) ? 0.f : ac * scale * z0;
  float r1    = ac * scale * z1;
  u_h[(size_t)node * 64 + lane]      = (half_t)r0;
  u_h[(size_t)node * 64 + lane + 32] = (half_t)r1;
}

// ---------------- K2: hyplin GEMM + nonlinearity + Performer feature GEMM ---
__global__ void __launch_bounds__(256) k2_hyplin(
    const half_t* __restrict__ u_h, const half_t* __restrict__ Wpack,
    const half_t* __restrict__ Ppack, half_t* __restrict__ qfeat,
    half_t* __restrict__ kfeat, half_t* __restrict__ vT) {
  int seg = blockIdx.y;
  int typ = seg >> 3;               // 0=q 1=k 2=v
  int h   = seg & 7;

  int wid   = threadIdx.x >> 5;
  int lane  = threadIdx.x & 31;
  int l15   = lane & 15;
  int lhalf = lane >> 4;
  int mbase = (blockIdx.x * 8 + wid) * 16;

  __shared__ __align__(16) float  muLDS[8][16][64];
  __shared__ __align__(16) half_t dsLDS[8][16][64];
  __shared__ float frontLDS[8][16];

  // ---- GEMM1: mu(16x64) = u(16x64) @ W^T ----
  v8f acc[4] = {};
  #pragma unroll
  for (int ks = 0; ks < 2; ++ks) {
    v16h a = load_a16(u_h + (size_t)(mbase + l15) * 64 + ks * 32 + 16 * lhalf);
    #pragma unroll
    for (int nt = 0; nt < 4; ++nt) {
      v16h b = *(const v16h*)(Wpack +
          ((((size_t)seg * 2 + ks) * 4 + nt) * 32 + lane) * 16);
      acc[nt] = wmma_f16(a, b, acc[nt]);
    }
  }
  #pragma unroll
  for (int nt = 0; nt < 4; ++nt)
    #pragma unroll
    for (int i = 0; i < 8; ++i)
      muLDS[wid][i + 8 * lhalf][nt * 16 + l15] = acc[nt][i];
  __syncthreads();

  // ---- row nonlinearity: proj(expmap0(mu)) ; lane pair per row ----
  {
    int r  = lane >> 1;
    int hs = lane & 1;
    const float* mrow = &muLDS[wid][r][0];
    float ss = 0.f;
    for (int c = hs * 32; c < hs * 32 + 32; ++c)
      if (c >= 1) { float x = mrow[c]; ss += x * x; }
    ss += __shfl_xor(ss, 1, 32);
    float xn    = fmaxf(sqrtf(ss), EPSF);
    float sh    = sinhf(xn);
    float scale = sh / xn;
    float ss2   = scale * scale * ss;
    float t     = sqrtf(1.0f + ss2);
    int n = mbase + r;
    if (typ == 2) {   // v: scatter [t, space...] into vT[h2][chan][col2]
      int h2   = n & 7;
      int col2 = h * (NN / 8) + (n >> 3);
      half_t* base = vT + ((size_t)h2 * 64) * NN + col2;
      for (int c = hs * 32; c < hs * 32 + 32; ++c) {
        float val = (c == 0) ? t : scale * mrow[c];
        base[(size_t)c * NN] = (half_t)val;
      }
    } else {          // q/k: stash ds (pos0 = 0) and front()
      for (int c = hs * 32; c < hs * 32 + 32; ++c) {
        float val = (c == 0) ? 0.f : scale * mrow[c];
        dsLDS[wid][r][c] = (half_t)val;
      }
      if (hs == 0)
        frontLDS[wid][r] = expf(1.0f - t * t) * INV_SQRT63;  // c=1, tau=0.5
    }
  }
  __syncthreads();
  if (typ == 2) return;

  // ---- GEMM2: pre(16x128) = ds(16x64) @ Ppad(64x128) ----
  v8f pacc[8] = {};
  #pragma unroll
  for (int ks = 0; ks < 2; ++ks) {
    v16h a = load_a16(&dsLDS[wid][l15][ks * 32 + 16 * lhalf]);
    #pragma unroll
    for (int nt = 0; nt < 8; ++nt) {
      v16h b = *(const v16h*)(Ppack + (((size_t)ks * 8 + nt) * 32 + lane) * 16);
      pacc[nt] = wmma_f16(a, b, pacc[nt]);
    }
  }
  // ---- exp + front, write features ----
  half_t* feat = (typ == 0) ? qfeat : kfeat;
  #pragma unroll
  for (int nt = 0; nt < 8; ++nt) {
    int nf = nt * 16 + l15;
    #pragma unroll
    for (int i = 0; i < 8; ++i) {
      int r = i + 8 * lhalf;
      float val = expf(pacc[nt][i] * SQRT2F) * frontLDS[wid][r];
      size_t g = (size_t)h * NN + (mbase + r);
      feat[g * NRF + nf] = (half_t)val;
    }
  }
}

// ---------------- K4: lastT[h2] (64x128) = V^T @ phi(k) ,  K = 32768 --------
// 1024 waves: wave = (h2, ntile, kchunk). B panel staged via TDM when present.
__global__ void __launch_bounds__(256) k4_last(const half_t* __restrict__ kfeat,
                                               const half_t* __restrict__ vT,
                                               float* __restrict__ lastT32) {
  int gwave  = (int)((blockIdx.x * blockDim.x + threadIdx.x) >> 5);
  int lane   = threadIdx.x & 31;
  int wid    = threadIdx.x >> 5;
  int kchunk = gwave & 15;
  int ntile  = (gwave >> 4) & 7;
  int h2     = gwave >> 7;
  int l15 = lane & 15, lhalf = lane >> 4;
  int nf = ntile * 16 + l15;
  int k0 = kchunk * 2048;
  v8f acc[4] = {};

#if USE_TDM
  __shared__ __align__(16) half_t stage[8][2][32 * 16];  // 32 rows x 16 feats
  // Issue a 2D TDM tile: 16 halves (dim0, contiguous) x 32 rows (dim1),
  // row stride = 8*128 halves = 1024 elements, data_size = 2B.
  auto issue = [&](int ks, int buf) {
    size_t ga = (size_t)kfeat +
        (((size_t)(k0 + ks * 32) * 8 + h2) * NRF + (size_t)ntile * 16) * 2;
    u32x4 g0;
    g0[0] = 1u;                                       // count=1, user mode
    g0[1] = (unsigned)(size_t)&stage[wid][buf][0];    // lds_addr
    g0[2] = (unsigned)ga;                             // global_addr[31:0]
    g0[3] = (unsigned)((ga >> 32) & 0x01FFFFFFu) | 0x80000000u;  // type=2
    i32x8 g1;
    g1[0] = 0x00010000;      // data_size=1 (2B), mask=0
    g1[1] = 16 << 16;        // tensor_dim0 = 16
    g1[2] = 32 << 16;        // tensor_dim1 = 32
    g1[3] = 16 << 16;        // tile_dim0 = 16
    g1[4] = 32;              // tile_dim1 = 32
    g1[5] = 1024;            // tensor_dim0_stride = 1024 elements
    g1[6] = 0;
    g1[7] = 0;
    i32x4 z4 = {};
#if __clang_major__ >= 23
    i32x8 z8 = {};
    __builtin_amdgcn_tensor_load_to_lds(g0, g1, z4, z4, z8, 0);
#else
    __builtin_amdgcn_tensor_load_to_lds(g0, g1, z4, z4, 0);
#endif
  };
  issue(0, 0);
  for (int ks = 0; ks < 64; ++ks) {
    int buf = ks & 1;
    if (ks < 63) {
      issue(ks + 1, buf ^ 1);
      __builtin_amdgcn_s_wait_tensorcnt(1);   // tile `ks` complete (in-order)
    } else {
      __builtin_amdgcn_s_wait_tensorcnt(0);
    }
    v16h b;
    #pragma unroll
    for (int j = 0; j < 16; ++j)
      b[j] = stage[wid][buf][(16 * lhalf + j) * 16 + l15];
    int kb = k0 + ks * 32;
    #pragma unroll
    for (int mt = 0; mt < 4; ++mt) {
      v16h a = load_a16(vT + ((size_t)h2 * 64 + mt * 16 + l15) * NN + kb +
                        16 * lhalf);
      acc[mt] = wmma_f16(a, b, acc[mt]);
    }
  }
#else
  (void)wid;
  for (int ks = 0; ks < 64; ++ks) {
    int kb = k0 + ks * 32;
    v16h b;
    #pragma unroll
    for (int j = 0; j < 16; ++j) {
      size_t g = (size_t)(kb + 16 * lhalf + j) * 8 + h2;
      b[j] = kfeat[g * NRF + nf];
    }
    #pragma unroll
    for (int mt = 0; mt < 4; ++mt) {
      v16h a = load_a16(vT + ((size_t)h2 * 64 + mt * 16 + l15) * NN + kb +
                        16 * lhalf);
      acc[mt] = wmma_f16(a, b, acc[mt]);
    }
  }
#endif

  #pragma unroll
  for (int mt = 0; mt < 4; ++mt)
    #pragma unroll
    for (int i = 0; i < 8; ++i) {
      int chan = mt * 16 + i + 8 * lhalf;
      atomicAdd(&lastT32[((size_t)h2 * 64 + chan) * NRF + nf], acc[mt][i]);
    }
}

__global__ void __launch_bounds__(256) k4b_cvt(const float* __restrict__ s,
                                               half_t* __restrict__ d) {
  int i = blockIdx.x * blockDim.x + threadIdx.x;
  if (i < 8 * 64 * 128) d[i] = (half_t)s[i];
}

// ---------------- K5: zo = phi(q) @ last, normalize, mean over h2 -----------
__global__ void __launch_bounds__(256) k5_zo(const half_t* __restrict__ qfeat,
                                             const half_t* __restrict__ lastT16,
                                             float* __restrict__ t_acc) {
  int gwave = (int)((blockIdx.x * blockDim.x + threadIdx.x) >> 5);
  int lane  = threadIdx.x & 31;
  int wid   = threadIdx.x >> 5;
  int mtile = gwave & 2047;
  int h2    = gwave >> 11;
  int l15 = lane & 15, lhalf = lane >> 4;
  __shared__ float zoLDS[8][16][64];

  v8f acc[4] = {};
  #pragma unroll
  for (int ks = 0; ks < 4; ++ks) {
    int kb = ks * 32;
    size_t g = (size_t)(mtile * 16 + l15) * 8 + h2;
    v16h a = load_a16(qfeat + g * NRF + kb + 16 * lhalf);
    #pragma unroll
    for (int nt = 0; nt < 4; ++nt) {
      v16h b = *(const v16h*)(lastT16 +
          ((size_t)h2 * 64 + nt * 16 + l15) * NRF + kb + 16 * lhalf);
      acc[nt] = wmma_f16(a, b, acc[nt]);
    }
  }
  #pragma unroll
  for (int nt = 0; nt < 4; ++nt)
    #pragma unroll
    for (int i = 0; i < 8; ++i)
      zoLDS[wid][i + 8 * lhalf][nt * 16 + l15] = acc[nt][i];
  __syncthreads();

  int r = lane >> 1, hs = lane & 1;
  const float* zr = &zoLDS[wid][r][0];
  float z0v = zr[0];
  float ss = 0.f;
  for (int c = hs * 32; c < hs * 32 + 32; ++c)
    if (c >= 1) { float x = zr[c]; ss += x * x; }
  ss += __shfl_xor(ss, 1, 32);
  float inv = 1.0f / sqrtf(fabsf(-z0v * z0v + ss));   // sqrt(c)=1
  int n2 = mtile * 16 + r;
  for (int c = hs * 32; c < hs * 32 + 32; ++c)
    atomicAdd(&t_acc[(size_t)n2 * 64 + c], zr[c] * inv * 0.125f);
}

// ---------------- K6: final Minkowski normalize -> z_next -------------------
__global__ void __launch_bounds__(256) k6_final(const float* __restrict__ t_acc,
                                                float* __restrict__ out) {
  int node = (int)((blockIdx.x * blockDim.x + threadIdx.x) >> 5);
  int lane = threadIdx.x & 31;
  const float* tr = t_acc + (size_t)node * 64;
  float a0 = tr[lane], a1 = tr[lane + 32];
  float s = ((lane == 0) ? -a0 * a0 : a0 * a0) + a1 * a1;
  #pragma unroll
  for (int m = 16; m; m >>= 1) s += __shfl_xor(s, m, 32);
  float inv = 1.0f / sqrtf(fabsf(s));
  out[(size_t)node * 64 + lane]      = a0 * inv;
  out[(size_t)node * 64 + lane + 32] = a1 * inv;
}

// ---------------- K7: edge link loss ----------------------------------------
__global__ void __launch_bounds__(256) k7_edges(const half_t* __restrict__ qfeat,
                                                const half_t* __restrict__ kfeat,
                                                const int* __restrict__ eidx,
                                                float* __restrict__ loss) {
  int tid = blockIdx.x * blockDim.x + threadIdx.x;   // (e, h2)
  int e = tid >> 3, h2 = tid & 7;
  int s = eidx[e];            // start
  int t = eidx[NE + e];       // end
  const half_t* qr = qfeat + ((size_t)t * 8 + h2) * NRF;
  const half_t* kr = kfeat + ((size_t)s * 8 + h2) * NRF;
  float acc = 0.f;
  #pragma unroll 8
  for (int i = 0; i < NRF; ++i) acc += (float)qr[i] * (float)kr[i];
  __shared__ float red[256];
  red[threadIdx.x] = acc;
  __syncthreads();
  for (int off = 128; off; off >>= 1) {
    if ((int)threadIdx.x < off) red[threadIdx.x] += red[threadIdx.x + off];
    __syncthreads();
  }
  if (threadIdx.x == 0)
    atomicAdd(loss, red[0] * (1.0f / (8.0f * (float)NE)));
}

// ---------------------------------------------------------------------------
extern "C" void kernel_launch(void* const* d_in, const int* in_sizes, int n_in,
                              void* d_out, int out_size, void* d_ws,
                              size_t ws_size, hipStream_t stream) {
  const float* z  = (const float*)d_in[0];
  const float* Wq = (const float*)d_in[1];
  const float* Wk = (const float*)d_in[2];
  const float* Wv = (const float*)d_in[3];
  const float* P  = (const float*)d_in[4];
  const int* eidx = (const int*)d_in[5];
  float* out = (float*)d_out;

  char* ws = (char*)d_ws;
  half_t* u_h     = (half_t*)(ws + OFF_U);
  half_t* qfeat   = (half_t*)(ws + OFF_QF);
  half_t* kfeat   = (half_t*)(ws + OFF_KF);
  half_t* vT      = (half_t*)(ws + OFF_VT);
  float*  lastT32 = (float*)(ws + OFF_L32);
  half_t* lastT16 = (half_t*)(ws + OFF_L16);
  float*  t_acc   = (float*)(ws + OFF_T);
  half_t* Wpack   = (half_t*)(ws + OFF_WP);
  half_t* Ppack   = (half_t*)(ws + OFF_PP);

  hipMemsetAsync(lastT32, 0, SZ_L32, stream);
  hipMemsetAsync(t_acc, 0, SZ_T, stream);
  hipMemsetAsync(out + (size_t)NN * 64, 0, sizeof(float), stream);

  k0_pack<<<416, 256, 0, stream>>>(Wq, Wk, Wv, P, Wpack, Ppack);
  k1_build_u<<<NN / 8, 256, 0, stream>>>(z, u_h);
  k2_hyplin<<<dim3(256, 24), 256, 0, stream>>>(u_h, Wpack, Ppack, qfeat, kfeat,
                                               vT);
  k4_last<<<128, 256, 0, stream>>>(kfeat, vT, lastT32);
  k4b_cvt<<<256, 256, 0, stream>>>(lastT32, lastT16);
  k5_zo<<<2048, 256, 0, stream>>>(qfeat, lastT16, t_acc);
  k6_final<<<NN / 8, 256, 0, stream>>>(t_acc, out);
  k7_edges<<<(NE * 8) / 256, 256, 0, stream>>>(qfeat, kfeat, eidx,
                                               out + (size_t)NN * 64);
}
